// GCN_enc_19963007992111
// MI455X (gfx1250) — matrix operand order repaired and checked by reading, not compiled, not run on hardware
//
#include <hip/hip_runtime.h>

#define N_NODES 100000
#define N_PAD   100096            // 782 * 128, padded row count for GEMM tiles
#define N_EDGES 1600000
#define DIM 128
#define EPS 1e-5f

typedef __attribute__((ext_vector_type(16))) _Float16 v16h;
typedef __attribute__((ext_vector_type(8)))  float    v8f;

// ---------------------------------------------------------------- utilities
__global__ void zero_kernel(float* __restrict__ p, long n) {
    long i = (long)blockIdx.x * blockDim.x + threadIdx.x;
    if (i < n) p[i] = 0.f;
}

// hbuf[0:N*D) = x, pad rows zeroed (run once per call before layer 0)
__global__ void copy_pad_kernel(const float* __restrict__ x,
                                float* __restrict__ hbuf) {
    long i = (long)blockIdx.x * blockDim.x + threadIdx.x;
    if (i < (long)N_PAD * DIM) {
        hbuf[i] = (i < (long)N_NODES * DIM) ? x[i] : 0.f;
    }
}

// One wave per edge; each lane handles 4 features (float4 gather + 4 atomics).
__global__ void scatter_add_kernel(const float* __restrict__ h,
                                   const int* __restrict__ src,
                                   const int* __restrict__ dst,
                                   float* __restrict__ agg) {
    long t = (long)blockIdx.x * blockDim.x + threadIdx.x;
    int e = (int)(t >> 5);
    if (e >= N_EDGES) return;
    int f4 = (int)(t & 31) * 4;
    int s = src[e];
    int d = dst[e];
    const float4 v = *(const float4*)(h + (long)s * DIM + f4);
    float* p = agg + (long)d * DIM + f4;
    atomicAdd(p + 0, v.x);
    atomicAdd(p + 1, v.y);
    atomicAdd(p + 2, v.z);
    atomicAdd(p + 3, v.w);
}

// ------------------------------------------------------------- WMMA GEMM
// A fragment (f16 16x32, MxK), contiguous K order:
//   lanes 0-15 : row=lane,    elems 0..15 -> K=k0..k0+15
//   lanes16-31 : row=lane-16, elems 0..15 -> K=k0+16..k0+31
__device__ inline v16h load_a_frag(const float* __restrict__ base, int row,
                                   int k0, int lane) {
    const float4* p4 =
        (const float4*)(base + (long)row * DIM + k0 + ((lane >> 4) * 16));
    float4 x0 = p4[0], x1 = p4[1];   // K = off .. off+7
    float4 x2 = p4[2], x3 = p4[3];   // K = off+8 .. off+15
    v16h a;
    a[0]  = (_Float16)x0.x; a[1]  = (_Float16)x0.y;
    a[2]  = (_Float16)x0.z; a[3]  = (_Float16)x0.w;
    a[4]  = (_Float16)x1.x; a[5]  = (_Float16)x1.y;
    a[6]  = (_Float16)x1.z; a[7]  = (_Float16)x1.w;
    a[8]  = (_Float16)x2.x; a[9]  = (_Float16)x2.y;
    a[10] = (_Float16)x2.z; a[11] = (_Float16)x2.w;
    a[12] = (_Float16)x3.x; a[13] = (_Float16)x3.y;
    a[14] = (_Float16)x3.z; a[15] = (_Float16)x3.w;
    return a;
}

// y = agg @ Wrel + h @ Wroot + b  (block = 128 padded rows, 8 waves, no guards)
__global__ void __launch_bounds__(256)
gemm_wmma_kernel(const float* __restrict__ agg, const float* __restrict__ h,
                 const float* __restrict__ Wrel, const float* __restrict__ Wroot,
                 const float* __restrict__ bias, float* __restrict__ y) {
    __shared__ _Float16 sWrel[DIM * DIM];    // transposed: [f][k], f16
    __shared__ _Float16 sWroot[DIM * DIM];

    int tid = threadIdx.x;
    for (int idx = tid; idx < DIM * DIM; idx += 256) {
        int k = idx >> 7, f = idx & (DIM - 1);
        sWrel[f * DIM + k]  = (_Float16)Wrel[idx];
        sWroot[f * DIM + k] = (_Float16)Wroot[idx];
    }
    __syncthreads();

    int wave    = tid >> 5;                  // 0..7
    int lane    = tid & 31;
    int rowbase = blockIdx.x * 128 + wave * 16;
    int arow    = rowbase + (lane & 15);

    v16h a_agg[4], a_h[4];                   // K = 128 -> 4 steps of 32
#pragma unroll
    for (int ks = 0; ks < 4; ks++) {
        a_agg[ks] = load_a_frag(agg, arow, ks * 32, lane);
        a_h[ks]   = load_a_frag(h,   arow, ks * 32, lane);
    }

    int col0  = lane & 15;
    int khalf = (lane >> 4) * 16;            // B layout: lanes16-31 hold K=16..31
    // C layout: VGPR r -> M = r (lanes 0-15) / r+8 (lanes 16-31), N = lane&15
    float* ybase = y + (long)(rowbase + (lane >> 4) * 8) * DIM + col0;

    for (int nt = 0; nt < 8; nt++) {
        int col  = nt * 16 + col0;
        float bv = bias[col];
        v8f c;
#pragma unroll
        for (int r = 0; r < 8; r++) c[r] = bv;

#pragma unroll
        for (int ks = 0; ks < 4; ks++) {
            v16h brel  = *(const v16h*)&sWrel[col * DIM + ks * 32 + khalf];
            c = __builtin_amdgcn_wmma_f32_16x16x32_f16(
                    false, a_agg[ks], false, brel, (short)0, c, false, false);
            v16h broot = *(const v16h*)&sWroot[col * DIM + ks * 32 + khalf];
            c = __builtin_amdgcn_wmma_f32_16x16x32_f16(
                    false, a_h[ks], false, broot, (short)0, c, false, false);
        }

        float* yp = ybase + nt * 16;
#pragma unroll
        for (int r = 0; r < 8; r++) {
            yp[r * DIM] = c[r];              // immediate-offset stores
        }
    }
}

// ------------------------------------------------------------- BatchNorm
__global__ void col_stats_kernel(const float* __restrict__ y,
                                 float* __restrict__ sums,
                                 float* __restrict__ sumsq) {
    __shared__ float ssum[256];
    __shared__ float ssq[256];
    int c    = threadIdx.x & (DIM - 1);
    int half = threadIdx.x >> 7;
    int rend = min(blockIdx.x * 128 + 128, N_NODES);
    float s = 0.f, q = 0.f;
    for (int r = blockIdx.x * 128 + half; r < rend; r += 2) {
        float v = y[(long)r * DIM + c];
        s += v;
        q += v * v;
    }
    ssum[threadIdx.x] = s;
    ssq[threadIdx.x]  = q;
    __syncthreads();
    if (threadIdx.x < DIM) {
        atomicAdd(&sums[c],  ssum[c] + ssum[c + 128]);
        atomicAdd(&sumsq[c], ssq[c]  + ssq[c + 128]);
    }
}

__global__ void scale_shift_kernel(const float* __restrict__ sums,
                                   const float* __restrict__ sumsq,
                                   const float* __restrict__ gamma,
                                   const float* __restrict__ beta,
                                   float* __restrict__ scale,
                                   float* __restrict__ shift) {
    int c = threadIdx.x;
    float m  = sums[c]  * (1.f / (float)N_NODES);
    float v  = sumsq[c] * (1.f / (float)N_NODES) - m * m;
    float sc = gamma[c] * rsqrtf(v + EPS);
    scale[c] = sc;
    shift[c] = beta[c] - m * sc;
}

__global__ void bn_relu_acc_kernel(const float* __restrict__ y,
                                   const float* __restrict__ scale,
                                   const float* __restrict__ shift,
                                   float* __restrict__ hout,
                                   float* __restrict__ tot, int first) {
    long i = (long)blockIdx.x * blockDim.x + threadIdx.x;
    if (i < (long)N_NODES * DIM) {
        int c   = (int)(i & (DIM - 1));
        float v = fmaxf(fmaf(y[i], scale[c], shift[c]), 0.f);
        hout[i] = v;
        tot[i]  = first ? v : tot[i] + v;
    }
}

// ------------------------------------------------------------- launcher
extern "C" void kernel_launch(void* const* d_in, const int* in_sizes, int n_in,
                              void* d_out, int out_size, void* d_ws, size_t ws_size,
                              hipStream_t stream) {
    (void)in_sizes; (void)n_in; (void)out_size; (void)ws_size;
    const float* x        = (const float*)d_in[0];
    const int*   edge     = (const int*)d_in[1];
    const int*   src      = edge;                // edge_index[0]
    const int*   dst      = edge + N_EDGES;      // edge_index[1]
    const float* Wrel_in  = (const float*)d_in[2];
    const float* Wroot_in = (const float*)d_in[3];
    const float* b_in     = (const float*)d_in[4];
    const float* g_in     = (const float*)d_in[5];
    const float* be_in    = (const float*)d_in[6];
    const float* Wrel_m   = (const float*)d_in[7];
    const float* Wroot_m  = (const float*)d_in[8];
    const float* b_m      = (const float*)d_in[9];
    const float* g_m      = (const float*)d_in[10];
    const float* be_m     = (const float*)d_in[11];

    const long ndp = (long)N_PAD * DIM;          // padded buffer size
    float* ws    = (float*)d_ws;
    float* agg   = ws;                // [N_PAD, D]
    float* y     = ws + ndp;          // [N_PAD, D] pre-BN output
    float* hbuf  = ws + 2 * ndp;      // [N_PAD, D] current activations
    float* sums  = ws + 3 * ndp;      // [D]
    float* sumsq = sums + DIM;        // [D]
    float* scale = sumsq + DIM;       // [D]
    float* shift = scale + DIM;       // [D]
    float* tot   = (float*)d_out;     // [N, D]

    const int pb = (int)((ndp + 255) / 256);
    const int zb = (int)(((long)N_NODES * DIM + 255) / 256);
    const int sb = (int)(((long)N_EDGES * 32 + 255) / 256);
    const int gb = N_PAD / 128;                  // 782

    // stage x into padded hbuf (pad rows zeroed); layers always read hbuf
    copy_pad_kernel<<<pb, 256, 0, stream>>>(x, hbuf);

    for (int layer = 0; layer < 4; layer++) {
        const float* Wrel  = (layer == 0) ? Wrel_in  : Wrel_m  + (long)(layer - 1) * DIM * DIM;
        const float* Wroot = (layer == 0) ? Wroot_in : Wroot_m + (long)(layer - 1) * DIM * DIM;
        const float* bias  = (layer == 0) ? b_in  : b_m  + (layer - 1) * DIM;
        const float* gamma = (layer == 0) ? g_in  : g_m  + (layer - 1) * DIM;
        const float* beta  = (layer == 0) ? be_in : be_m + (layer - 1) * DIM;

        zero_kernel<<<pb, 256, 0, stream>>>(agg, ndp);
        zero_kernel<<<1, 256, 0, stream>>>(sums, 2 * DIM);
        scatter_add_kernel<<<sb, 256, 0, stream>>>(hbuf, src, dst, agg);
        gemm_wmma_kernel<<<gb, 256, 0, stream>>>(agg, hbuf, Wrel, Wroot, bias, y);
        col_stats_kernel<<<gb, 256, 0, stream>>>(y, sums, sumsq);
        scale_shift_kernel<<<1, DIM, 0, stream>>>(sums, sumsq, gamma, beta, scale, shift);
        bn_relu_acc_kernel<<<zb, 256, 0, stream>>>(y, scale, shift, hbuf, tot,
                                                   layer == 0 ? 1 : 0);
    }
}
